// VisionEncoder_13082470383712
// MI455X (gfx1250) — matrix operand order repaired and checked by reading, not compiled, not run on hardware
//
#include <hip/hip_runtime.h>

typedef __attribute__((ext_vector_type(16))) __bf16 v16bf;
typedef __attribute__((ext_vector_type(8)))  float  v8f;

#define HW    4096          // 64*64 spatial positions
#define CF    1536          // fused channels
#define KT    48            // CF / 32 K-tiles
#define OC    512           // output channels
#define NB    16            // batch
#define NBOX  10

// workspace layout (bytes)
#define OFF_WPK   0UL
#define OFF_FUSED (2UL*1024*1024)                                   // 2MB aligned
#define OFF_ATTR  (OFF_FUSED + (size_t)NB*KT*HW*32*2)               // after 201MB bf16

// ---------------------------------------------------------------------------
// CDNA5 async global->LDS 16-byte copy via inline asm (GLOBAL_LOAD_ASYNC_TO_LDS_B128).
// VDST = VGPR with LDS byte address, VADDR = 64-bit global address, SADDR = off.
// Tracked by ASYNCcnt; wait with s_wait_asynccnt.
__device__ __forceinline__ void async_copy16(const void* g, void* l) {
    unsigned lds_addr =
        (unsigned)(size_t)(__attribute__((address_space(3))) char*)l;
    asm volatile("global_load_async_to_lds_b128 %0, %1, off"
                 :
                 : "v"(lds_addr), "v"(g)
                 : "memory");
}

__device__ __forceinline__ void async_wait0() {
    asm volatile("s_wait_asynccnt 0" ::: "memory");
}

// ---------------------------------------------------------------------------
// boxes passthrough + W fp32->bf16, pre-packed into the WMMA A-fragment
// per-lane layout: Wpack[kt][mt][lane][e], lane 0-15 -> rows, lhi -> K-half,
// K=kbase+16*lhi+e   (contiguous K order, matches B fragment)
__global__ __launch_bounds__(256)
void k_misc(const float* __restrict__ boxes, float* __restrict__ out_boxes,
            const float* __restrict__ wconv, __bf16* __restrict__ Wpack)
{
    int i = blockIdx.x * blockDim.x + threadIdx.x;
    if (i < NB*NBOX*4) out_boxes[i] = boxes[i];
    if (i < OC*CF) {
        int e    = i & 15;
        int lane = (i >> 4) & 31;
        int mt   = (i >> 9) & 31;
        int kt   = i >> 14;
        int o     = mt*16 + (lane & 15);
        int kbase = (lane >> 4) * 16;
        int k     = kt*32 + kbase + e;
        Wpack[i] = (__bf16)wconv[o*CF + k];
    }
}

// ---------------------------------------------------------------------------
// fused = concat(c3, bilinear2x(c4)) -> bf16, packed as [b][kt][hw][k%32]
__global__ __launch_bounds__(256)
void k_pack_fused(const float* __restrict__ c3, const float* __restrict__ c4,
                  __bf16* __restrict__ fusedT)
{
    size_t idx = (size_t)blockIdx.x * blockDim.x + threadIdx.x;
    const size_t total = (size_t)NB * CF * HW;
    if (idx >= total) return;
    int hw = (int)(idx % HW);
    int c  = (int)((idx / HW) % CF);
    int b  = (int)(idx / ((size_t)CF * HW));

    float val;
    if (c < 512) {
        val = c3[((size_t)(b*512 + c))*HW + hw];
    } else {
        // half-pixel bilinear 2x: src = dst/2 - 0.25
        int cc = c - 512;
        int h = hw >> 6, w = hw & 63;
        float sy = h*0.5f - 0.25f, sx = w*0.5f - 0.25f;
        int y0 = (int)floorf(sy), x0 = (int)floorf(sx);
        float fy = sy - (float)y0, fx = sx - (float)x0;
        int y0c = max(0, min(31, y0)), y1c = max(0, min(31, y0 + 1));
        int x0c = max(0, min(31, x0)), x1c = max(0, min(31, x0 + 1));
        const float* p = c4 + ((size_t)(b*1024 + cc))*1024;
        float v00 = p[y0c*32 + x0c], v01 = p[y0c*32 + x1c];
        float v10 = p[y1c*32 + x0c], v11 = p[y1c*32 + x1c];
        val = (1.f-fy)*((1.f-fx)*v00 + fx*v01) + fy*((1.f-fx)*v10 + fx*v11);
    }
    fusedT[(((size_t)(b*KT + (c >> 5))*HW) + hw)*32 + (c & 31)] = (__bf16)val;
}

// ---------------------------------------------------------------------------
// GEMM: attr[b,o,hw] = relu(BN(sum_c W[o,c]*fused[b,c,hw] + bias))
// 128x128 block tile, 8 waves (2M x 4N), wave = 64x32 = 4x2 wmma tiles.
// A/B K-step tiles (8KB each, contiguous in the packed layouts) are staged
// into LDS with double-buffered async global->LDS copies.
__global__ __launch_bounds__(256)
void k_gemm(const __bf16* __restrict__ Wpack, const __bf16* __restrict__ fusedT,
            const float* __restrict__ bconv, const float* __restrict__ gamma,
            const float* __restrict__ beta,  const float* __restrict__ mean,
            const float* __restrict__ var,   float* __restrict__ attr)
{
    __shared__ __align__(128) char smem[2][16384];   // [buf][A:8KB | B:8KB]

    const int tid  = threadIdx.x;
    const int lane = tid & 31;
    const int wave = tid >> 5;
    const int b    = blockIdx.z;
    const int mt0  = blockIdx.y * 8;                 // first 16-row mtile
    const int nblk = blockIdx.x * 128;               // first column
    const int wm   = wave & 1;                       // 0..1 : 64-row half
    const int wn   = wave >> 1;                      // 0..3 : 32-col slice
    const int l15  = lane & 15;
    const int lhi  = lane >> 4;

    const char* gA0 = (const char*)Wpack;
    const char* gB0 = (const char*)fusedT;

    // stage K-tile kt into buffer buf (16KB, 64B per thread)
    auto stage = [&](int buf, int kt) {
        const char* gA = gA0 + ((size_t)kt*16384 + (size_t)mt0*512)*2;
        const char* gB = gB0 + (((size_t)(b*KT + kt)*HW) + nblk)*64;
        char* la = smem[buf];
        char* lb = smem[buf] + 8192;
        async_copy16(gA + tid*32,      la + tid*32);
        async_copy16(gA + tid*32 + 16, la + tid*32 + 16);
        async_copy16(gB + tid*32,      lb + tid*32);
        async_copy16(gB + tid*32 + 16, lb + tid*32 + 16);
    };

    v8f acc[4][2] = {};

    stage(0, 0);
    for (int kt = 0; kt < KT; ++kt) {
        async_wait0();
        __syncthreads();                  // buf[kt&1] ready for every wave
        if (kt + 1 < KT) stage((kt + 1) & 1, kt + 1);
        if (kt + 2 < KT)                  // pull kt+2 B toward L2 (global_prefetch_b8)
            __builtin_prefetch(gB0 + (((size_t)(b*KT + kt + 2)*HW) + nblk)*64
                               + tid*32, 0, 1);

        const __bf16* A  = (const __bf16*)(smem[kt & 1]);
        const __bf16* Bt = (const __bf16*)(smem[kt & 1] + 8192);

        v16bf bfrag[2];
        #pragma unroll
        for (int nf = 0; nf < 2; ++nf)
            bfrag[nf] = *(const v16bf*)(Bt + (size_t)(wn*32 + nf*16 + l15)*32
                                           + lhi*16);
        #pragma unroll
        for (int mf = 0; mf < 4; ++mf) {
            v16bf afrag = *(const v16bf*)(A + (size_t)((wm*4 + mf)*32 + lane)*16);
            #pragma unroll
            for (int nf = 0; nf < 2; ++nf)
                acc[mf][nf] = __builtin_amdgcn_wmma_f32_16x16x32_bf16(
                    false, afrag, false, bfrag[nf], (short)0, acc[mf][nf],
                    false, false);
        }
        __syncthreads();                  // all reads of buf done before reuse
    }

    // epilogue: fused conv-bias + eval BN + relu
    // C/D layout: VGPR r -> row r + 8*lhi, col = l15
    #pragma unroll
    for (int mf = 0; mf < 4; ++mf) {
        #pragma unroll
        for (int r = 0; r < 8; ++r) {
            int o = (mt0 + wm*4 + mf)*16 + lhi*8 + r;
            float inv = gamma[o] * rsqrtf(var[o] + 1e-5f);
            float sh  = bconv[o]*inv + beta[o] - mean[o]*inv;
            #pragma unroll
            for (int nf = 0; nf < 2; ++nf) {
                int hw = nblk + wn*32 + nf*16 + l15;
                float v = acc[mf][nf][r]*inv + sh;
                attr[((size_t)(b*OC + o))*HW + hw] = v > 0.f ? v : 0.f;
            }
        }
    }
}

// ---------------------------------------------------------------------------
// ROI adaptive-avg-pool 5x5, exact _bin_weights integer math
__global__ __launch_bounds__(256)
void k_pool(const float* __restrict__ attr, const float* __restrict__ boxes,
            float* __restrict__ obj)
{
    int bb  = blockIdx.x;              // 0..159
    int b   = bb / NBOX, box = bb % NBOX;
    const float* bx = boxes + (size_t)(b*NBOX + box)*4;
    const float s = 64.f / 512.f;
    int x1 = min(63, max(0, (int)floorf(bx[0]*s)));
    int y1 = min(63, max(0, (int)floorf(bx[1]*s)));
    int x2 = min(63, max(0, (int)ceilf (bx[2]*s)));
    int y2 = min(63, max(0, (int)ceilf (bx[3]*s)));
    if (x2 <= x1) x2 = min(x1 + 1, 64);
    if (y2 <= y1) y2 = min(y1 + 1, 64);
    int Lx = x2 - x1, Ly = y2 - y1;
    int xs[5], xe[5], ys[5], ye[5];
    #pragma unroll
    for (int k = 0; k < 5; ++k) {
        xs[k] = x1 + (k*Lx)/5;  xe[k] = x1 + ((k+1)*Lx + 4)/5;
        ys[k] = y1 + (k*Ly)/5;  ye[k] = y1 + ((k+1)*Ly + 4)/5;
    }
    for (int c = threadIdx.x; c < OC; c += blockDim.x) {
        const float* f = attr + ((size_t)(b*OC + c))*HW;
        #pragma unroll
        for (int p = 0; p < 5; ++p) {
            #pragma unroll
            for (int q = 0; q < 5; ++q) {
                float acc = 0.f;
                for (int h = ys[p]; h < ye[p]; ++h)
                    for (int w = xs[q]; w < xe[q]; ++w)
                        acc += f[h*64 + w];
                float wgt = 1.f / ((float)(ye[p]-ys[p]) * (float)(xe[q]-xs[q]));
                obj[(((size_t)(b*NBOX + box)*OC + c)*25) + p*5 + q] = acc * wgt;
            }
        }
    }
}

// ---------------------------------------------------------------------------
extern "C" void kernel_launch(void* const* d_in, const int* in_sizes, int n_in,
                              void* d_out, int out_size, void* d_ws, size_t ws_size,
                              hipStream_t stream)
{
    const float* c3    = (const float*)d_in[0];
    const float* c4    = (const float*)d_in[1];
    const float* boxes = (const float*)d_in[2];
    const float* wconv = (const float*)d_in[3];
    const float* bconv = (const float*)d_in[4];
    const float* gamma = (const float*)d_in[5];
    const float* beta  = (const float*)d_in[6];
    const float* mean  = (const float*)d_in[7];
    const float* var   = (const float*)d_in[8];

    char* ws = (char*)d_ws;
    __bf16* Wpack  = (__bf16*)(ws + OFF_WPK);
    __bf16* fusedT = (__bf16*)(ws + OFF_FUSED);
    float*  attr   = (float*)(ws + OFF_ATTR);
    float*  out_boxes = (float*)d_out;
    float*  obj       = (float*)d_out + NB*NBOX*4;

    k_misc<<<(OC*CF + 255)/256, 256, 0, stream>>>(boxes, out_boxes, wconv, Wpack);

    size_t totalF = (size_t)NB * CF * HW;
    k_pack_fused<<<(unsigned)((totalF + 255)/256), 256, 0, stream>>>(c3, c4, fusedT);

    dim3 g(HW/128, OC/128, NB);   // 32 x 4 x 16
    k_gemm<<<g, 256, 0, stream>>>(Wpack, fusedT, bconv, gamma, beta, mean, var, attr);

    k_pool<<<NB*NBOX, 256, 0, stream>>>(attr, boxes, obj);
}
